// MPNN_72378788872769
// MI455X (gfx1250) — matrix operand order repaired and checked by reading, not compiled, not run on hardware
//
#include <hip/hip_runtime.h>
#include <hip/hip_bf16.h>
#include <math.h>

// ---------------------------------------------------------------------------
// MPNN: B=4, N=64, F_IN=32, E_DIM=32, F=64, T=3
//
// Pipeline:
//   1) embed_kernel : H = relu(X @ W_embed + b)                     [fp32 VALU]
//   2) edge_amask   : Amask[b][(n,i)][(m,j)] =
//        relu(E[b,nm] @ W_edge + b_edge) * mask[b,nm]   (bf16, 134 MB -> L2)
//        GEMM [16384 x 32] @ [32 x 4096] via v_wmma_f32_16x16x32_bf16
//   3) per round: gemv_kernel  agg_b = M_b @ h_b  (bandwidth-bound, from L2)
//                 gru_kernel   2-step fused GRU   (fp32 VALU, tiny)
// ---------------------------------------------------------------------------

typedef __attribute__((ext_vector_type(16))) __bf16 v16bf;
typedef __attribute__((ext_vector_type(8)))  float  v8f;

__device__ __forceinline__ unsigned short f2bf_bits(float f) {
  unsigned u = __builtin_bit_cast(unsigned, f);
  u += 0x7FFFu + ((u >> 16) & 1u);          // round-to-nearest-even
  return (unsigned short)(u >> 16);
}
__device__ __forceinline__ __bf16 bits2bf(unsigned short h) {
  return __builtin_bit_cast(__bf16, h);
}
__device__ __forceinline__ float bflo2f(unsigned u) {
  return __builtin_bit_cast(float, u << 16);
}
__device__ __forceinline__ float bfhi2f(unsigned u) {
  return __builtin_bit_cast(float, u & 0xffff0000u);
}

// ---------------------------------------------------------------------------
// 1) Embedding: H[row,f] = relu(X[row,:] @ W_embed[:,f] + b_embed[f])
//    rows = B*N = 256, f = 64  -> 16384 threads
// ---------------------------------------------------------------------------
__global__ __launch_bounds__(256) void embed_kernel(
    const float* __restrict__ X, const float* __restrict__ W,
    const float* __restrict__ bias, float* __restrict__ H) {
  const int idx = blockIdx.x * 256 + threadIdx.x;   // 0..16383
  const int rowN = idx >> 6, f = idx & 63;
  float acc = bias[f];
#pragma unroll
  for (int e = 0; e < 32; ++e) acc += X[rowN * 32 + e] * W[e * 64 + f];
  H[idx] = fmaxf(acc, 0.f);
}

// ---------------------------------------------------------------------------
// 2) Edge MLP as WMMA GEMM with fused (bias, relu, mask, transpose-store).
//    Grid: x = ff-quarter (4 x 1024 cols), y = 128-edge groups (128 blocks).
//    Block: 256 threads = 8 waves; wave owns 16 edges, loops 64 ff tiles.
// ---------------------------------------------------------------------------
#define FFBLK 1024

__global__ __launch_bounds__(256) void edge_amask_kernel(
    const float* __restrict__ E, const float* __restrict__ W_edge,
    const float* __restrict__ b_edge, const float* __restrict__ mask_in,
    unsigned short* __restrict__ Aout) {
  __shared__ unsigned short lds_w[FFBLK * 32];  // W_edge tile, bf16, [ff][K]
  __shared__ float lds_bias[FFBLK];

  const int tid = threadIdx.x;
  const int ffBase = blockIdx.x * FFBLK;

  // Stage W_edge quarter into LDS as bf16, [ff][K] so B-fragment reads are
  // two 8-halfword runs per lane.
  for (int idx = tid; idx < FFBLK * 32; idx += 256) {
    const int K = idx / FFBLK, ff = idx - K * FFBLK;  // coalesced global read
    lds_w[ff * 32 + K] = f2bf_bits(W_edge[K * 4096 + ffBase + ff]);
  }
  for (int idx = tid; idx < FFBLK; idx += 256)
    lds_bias[idx] = b_edge[ffBase + idx];
  __syncthreads();

  const int wave = tid >> 5, lane = tid & 31;
  const int hi = lane >> 4, lrow = lane & 15;
  const int edge0 = blockIdx.y * 128 + wave * 16;   // 16 consecutive edges
  const int myEdge = edge0 + lrow;                  // A-matrix row for lane
  const int b = edge0 >> 12;
  const int nm = edge0 & 4095;
  const int n = nm >> 6, mBase = nm & 63;           // fixed n per 16-edge tile

  // --- A fragment: 16x32 bf16 (ISA layout: lane=row; K 0-7/16-23 for lanes
  //     0-15, K 8-15/24-31 for lanes 16-31), K = E_DIM = 32 in one WMMA.
  v16bf afrag;
  {
    const float4* ep = reinterpret_cast<const float4*>(E + (size_t)myEdge * 32);
    const int blo = hi ? 2 : 0;   // float4 index of K base_lo (0 or 8)
    const int bhi = hi ? 6 : 4;   // float4 index of K base_hi (16 or 24)
    const float4 p0 = ep[blo], p1 = ep[blo + 1];
    const float4 p2 = ep[bhi], p3 = ep[bhi + 1];
    const float v[16] = {p0.x, p0.y, p0.z, p0.w, p1.x, p1.y, p1.z, p1.w,
                         p2.x, p2.y, p2.z, p2.w, p3.x, p3.y, p3.z, p3.w};
#pragma unroll
    for (int e = 0; e < 16; ++e) afrag[e] = bits2bf(f2bf_bits(v[e]));
  }

  // Adjacency mask per D-row (D row M = r + 8*hi -> edge edge0 + M).
  float maskf[8];
  {
    const float4* mp =
        reinterpret_cast<const float4*>(mask_in + edge0 + 8 * hi);
    const float4 m0 = mp[0], m1 = mp[1];
    maskf[0] = m0.x; maskf[1] = m0.y; maskf[2] = m0.z; maskf[3] = m0.w;
    maskf[4] = m1.x; maskf[5] = m1.y; maskf[6] = m1.z; maskf[7] = m1.w;
  }

  const size_t bOff = ((size_t)b) << 24;  // b * 4096 * 4096

  for (int t = 0; t < FFBLK / 16; ++t) {
    const int ffl = t * 16 + lrow;        // this lane's local column
    // --- B fragment: 32x16 bf16 (lane=column; lanes 0-15 hold K 0-7/16-23,
    //     lanes 16-31 hold K 8-15/24-31) — two 8-bf16 runs from LDS.
    v16bf bfrag;
    const unsigned short* bp = &lds_w[ffl * 32 + hi * 8];
#pragma unroll
    for (int e = 0; e < 16; ++e) bfrag[e] = bits2bf(bp[e < 8 ? e : e + 8]);

    v8f c = {};
    c = __builtin_amdgcn_wmma_f32_16x16x32_bf16(
        /*neg_a=*/false, afrag, /*neg_b=*/false, bfrag,
        /*c_mod=*/(short)0, c, /*reuse_a=*/false, /*reuse_b=*/false);

    const float biasv = lds_bias[ffl];
    const int ffg = ffBase + ffl;         // global ff = i*64 + j
    const int i = ffg >> 6, j = ffg & 63; // i fixed across the tile
    // Destination: Amask[b][(n*64 + i)][(m)*64 + j], m = mBase + r + 8*hi
    size_t dst = bOff + (size_t)(n * 64 + i) * 4096 +
                 (size_t)(mBase + 8 * hi) * 64 + (size_t)j;
#pragma unroll
    for (int r = 0; r < 8; ++r) {
      const float v = fmaxf(c[r] + biasv, 0.f) * maskf[r];
      Aout[dst + (size_t)r * 64] = f2bf_bits(v);
    }
  }
}

// ---------------------------------------------------------------------------
// 3a) Per-round aggregation: agg[b*4096 + r] = sum_k Amask[b][r][k] * H[b][k]
//     Wave-per-row GEMV; H_b staged in LDS; bandwidth-bound from L2.
// ---------------------------------------------------------------------------
__global__ __launch_bounds__(256) void gemv_kernel(
    const unsigned short* __restrict__ Amask, const float* __restrict__ H,
    float* __restrict__ agg) {
  __shared__ __align__(16) float hs[4096];
  const int tid = threadIdx.x;
  const int rowBase = blockIdx.x * 8;     // 8 waves -> 8 rows, same batch b
  const int b = rowBase >> 12;
  for (int idx = tid; idx < 4096; idx += 256) hs[idx] = H[b * 4096 + idx];
  __syncthreads();

  const int wave = tid >> 5, lane = tid & 31;
  const int row = rowBase + wave;
  const uint4* rp =
      reinterpret_cast<const uint4*>(Amask + (size_t)row * 4096);

  float acc = 0.f;
#pragma unroll 4
  for (int t = 0; t < 16; ++t) {
    const int eb = t * 256 + lane * 8;    // wave reads 512B contiguous / iter
    const uint4 q = rp[eb >> 3];
    if (t + 4 < 16)
      __builtin_prefetch(&rp[(eb + 4 * 256) >> 3], 0, 0);  // global_prefetch_b8
    const float4 h0 = *reinterpret_cast<const float4*>(&hs[eb]);
    const float4 h1 = *reinterpret_cast<const float4*>(&hs[eb + 4]);
    acc += bflo2f(q.x) * h0.x + bfhi2f(q.x) * h0.y +
           bflo2f(q.y) * h0.z + bfhi2f(q.y) * h0.w +
           bflo2f(q.z) * h1.x + bfhi2f(q.z) * h1.y +
           bflo2f(q.w) * h1.z + bfhi2f(q.w) * h1.w;
  }
#pragma unroll
  for (int off = 16; off > 0; off >>= 1) acc += __shfl_xor(acc, off, 32);
  if (lane == 0) agg[row] = acc;
}

// ---------------------------------------------------------------------------
// 3b) Fused 2-step Keras GRU (reset_after), h0 = 0. One block per (b,n) row.
// ---------------------------------------------------------------------------
__global__ __launch_bounds__(64) void gru_kernel(
    const float* __restrict__ Hin, const float* __restrict__ agg,
    const float* __restrict__ Wk, const float* __restrict__ Uk,
    const float* __restrict__ bk, const float* __restrict__ rbk,
    float* __restrict__ Hout) {
  __shared__ float x1[64], x2[64], h1[64];
  const int row = blockIdx.x, f = threadIdx.x;
  x1[f] = Hin[row * 64 + f];
  x2[f] = agg[row * 64 + f];
  __syncthreads();

  // step 1: x = H, h = 0  =>  hg = recurrent bias only
  float xz = bk[f], xr = bk[64 + f], xh = bk[128 + f];
  for (int e = 0; e < 64; ++e) {
    const float xv = x1[e];
    xz += xv * Wk[e * 192 + f];
    xr += xv * Wk[e * 192 + 64 + f];
    xh += xv * Wk[e * 192 + 128 + f];
  }
  const float z1 = 1.f / (1.f + expf(-(xz + rbk[f])));
  const float r1 = 1.f / (1.f + expf(-(xr + rbk[64 + f])));
  const float c1 = tanhf(xh + r1 * rbk[128 + f]);
  const float h1v = (1.f - z1) * c1;      // z1*h0 + (1-z1)*c1, h0 = 0
  h1[f] = h1v;
  __syncthreads();

  // step 2: x = agg, h = h1
  float xz2 = bk[f], xr2 = bk[64 + f], xh2 = bk[128 + f];
  float hz2 = rbk[f], hr2 = rbk[64 + f], hh2 = rbk[128 + f];
  for (int e = 0; e < 64; ++e) {
    const float xv = x2[e], hv = h1[e];
    xz2 += xv * Wk[e * 192 + f];        hz2 += hv * Uk[e * 192 + f];
    xr2 += xv * Wk[e * 192 + 64 + f];   hr2 += hv * Uk[e * 192 + 64 + f];
    xh2 += xv * Wk[e * 192 + 128 + f];  hh2 += hv * Uk[e * 192 + 128 + f];
  }
  const float z2 = 1.f / (1.f + expf(-(xz2 + hz2)));
  const float r2 = 1.f / (1.f + expf(-(xr2 + hr2)));
  const float c2 = tanhf(xh2 + r2 * hh2);
  Hout[row * 64 + f] = z2 * h1v + (1.f - z2) * c2;
}

// ---------------------------------------------------------------------------
// Launch
// ---------------------------------------------------------------------------
extern "C" void kernel_launch(void* const* d_in, const int* in_sizes, int n_in,
                              void* d_out, int out_size, void* d_ws,
                              size_t ws_size, hipStream_t stream) {
  (void)in_sizes; (void)n_in; (void)out_size; (void)ws_size;
  const float* X       = (const float*)d_in[0];
  const float* A       = (const float*)d_in[1];   // adjacency mask [B,N*N,1]
  const float* E       = (const float*)d_in[2];
  const float* W_embed = (const float*)d_in[3];
  const float* b_embed = (const float*)d_in[4];
  const float* W_edge  = (const float*)d_in[5];
  const float* b_edge  = (const float*)d_in[6];
  const float* gru_k   = (const float*)d_in[7];
  const float* gru_rk  = (const float*)d_in[8];
  const float* gru_b   = (const float*)d_in[9];
  const float* gru_rb  = (const float*)d_in[10];
  float* out = (float*)d_out;

  // Workspace layout: Amask bf16 (134,217,728 B) | H (65,536 B) | agg (65,536 B)
  char* ws = (char*)d_ws;
  unsigned short* Amask = (unsigned short*)ws;
  float* Hbuf = (float*)(ws + (size_t)134217728);
  float* aggb = (float*)(ws + (size_t)134217728 + 65536);

  embed_kernel<<<64, 256, 0, stream>>>(X, W_embed, b_embed, Hbuf);
  edge_amask_kernel<<<dim3(4, 128), 256, 0, stream>>>(E, W_edge, b_edge, A,
                                                      Amask);
  for (int t = 0; t < 3; ++t) {
    gemv_kernel<<<2048, 256, 0, stream>>>(Amask, Hbuf, aggb);
    gru_kernel<<<256, 64, 0, stream>>>(Hbuf, aggb, gru_k, gru_rk, gru_b,
                                       gru_rb, (t == 2) ? out : Hbuf);
  }
}
